// MambaMixer_3951369913007
// MI455X (gfx1250) — matrix operand order repaired and checked by reading, not compiled, not run on hardware
//
#include <hip/hip_runtime.h>
#include <hip/hip_bf16.h>
#include <math.h>

// ---------------- problem constants ----------------
#define BB     2
#define LL     2048
#define HH     768
#define II     1536
#define J2     3072          // 2*I
#define NN     16
#define DTR    48
#define KK     4
#define ML     (BB*LL)       // 4096 rows for all GEMMs (b,l flattened)

typedef __attribute__((ext_vector_type(16))) __bf16 v16bf;
typedef __attribute__((ext_vector_type(8)))  __bf16 v8bf;
typedef __attribute__((ext_vector_type(8)))  float  v8f;

// ---------------- workspace layout (all region sizes multiple of 256B) ----
#define OFF_ABF   ((size_t)0)                       // bf16 input   [4096][768]
#define OFF_W1T   (OFF_ABF  + (size_t)ML*HH*2)      // bf16 in_proj^T [3072][768]
#define OFF_XT    (OFF_W1T  + (size_t)J2*HH*2)      // bf16 x_proj^T  [80][1536]
#define OFF_OT    (OFF_XT   + (size_t)80*II*2)      // bf16 out_proj^T[768][1536]
#define OFF_PROJ  (OFF_OT   + (size_t)HH*II*2)      // f32 proj [B][3072][2048]
#define OFF_HID   (OFF_PROJ + (size_t)BB*J2*LL*4)   // f32 hidden [B][I][L]
#define OFF_HBF   (OFF_HID  + (size_t)BB*II*LL*4)   // bf16 hidden^T [4096][1536]
#define OFF_SSM   (OFF_HBF  + (size_t)ML*II*2)      // f32 ssm_p [4096][80]
#define OFF_DTV   (OFF_SSM  + (size_t)ML*80*4)      // f32 dt [4096][1536]
#define OFF_YBF   (OFF_DTV  + (size_t)ML*II*4)      // bf16 y [4096][1536]

// ---------------- prep kernels ----------------
__global__ void cvt_bf16_kernel(const float* __restrict__ src,
                                __bf16* __restrict__ dst, int n) {
    int idx = blockIdx.x * 256 + threadIdx.x;
    if (idx < n) dst[idx] = (__bf16)src[idx];
}

// src[rows][cols] f32  ->  dst[cols][rows] bf16
__global__ void cvt_transpose_kernel(const float* __restrict__ src,
                                     __bf16* __restrict__ dst,
                                     int rows, int cols) {
    int idx = blockIdx.x * 256 + threadIdx.x;
    if (idx >= rows * cols) return;
    int r = idx / cols, c = idx % cols;
    dst[(size_t)c * rows + r] = (__bf16)src[idx];
}

// ---------------- bf16 WMMA fragment loaders ----------------
// A: row-major [m][k]; 16-bit A layout: lane m=lane&15, K-halves via lane>>4
__device__ __forceinline__ v16bf load_frag_a(const __bf16* __restrict__ A,
                                             int ld, int row, int k0, int lane) {
    int kb = (lane >> 4) * 8;
    const __bf16* p = A + (size_t)row * ld + k0 + kb;
    v8bf lo = *(const v8bf*)(p);
    v8bf hi = *(const v8bf*)(p + 16);
    v16bf r;
#pragma unroll
    for (int e = 0; e < 8; ++e) { r[e] = lo[e]; r[e + 8] = hi[e]; }
    return r;
}

// B^T: row-major [n][k]: lane n=lane&15, K-halves via lane>>4 (same order as A)
__device__ __forceinline__ v16bf load_frag_b(const __bf16* __restrict__ Bt,
                                             int ld, int col, int k0, int lane) {
    int kb = (lane >> 4) * 8;
    const __bf16* p = Bt + (size_t)col * ld + k0 + kb;
    v8bf lo = *(const v8bf*)(p);
    v8bf hi = *(const v8bf*)(p + 16);
    v16bf r;
#pragma unroll
    for (int e = 0; e < 8; ++e) { r[e] = lo[e]; r[e + 8] = hi[e]; }
    return r;
}

// ======= register-blocked GEMM: 32Mx64N per wave (2x4 frags, 8 WMMA / 12 loads)
// block = 8 waves arranged 2x4 -> 64M x 256N macro-tile.
// mode 0: C row-major [M][Nn];  mode 1: in_proj store -> proj[b][j][l].
#define WMF 2
#define WNF 4
__global__ void gemm_bf16_wmma_blocked(const __bf16* __restrict__ A,
                                       const __bf16* __restrict__ Bt,
                                       float* __restrict__ C,
                                       int M, int Nn, int Kk, int mode) {
    int lane = threadIdx.x & 31;
    int wave = threadIdx.x >> 5;
    int wm = wave >> 2, wn = wave & 3;
    int nbn = Nn >> 8;                       // 256-wide N blocks
    int bm = blockIdx.x / nbn;
    int bn = blockIdx.x - bm * nbn;
    int m0 = bm * 64 + wm * 32;
    int n0 = bn * 256 + wn * 64;
    int r16 = lane & 15;

    v8f acc[WMF][WNF] = {};
    for (int k0 = 0; k0 < Kk; k0 += 32) {
        v16bf af[WMF];
#pragma unroll
        for (int im = 0; im < WMF; ++im)
            af[im] = load_frag_a(A, Kk, m0 + im * 16 + r16, k0, lane);
        v16bf bf[WNF];
#pragma unroll
        for (int in = 0; in < WNF; ++in)
            bf[in] = load_frag_b(Bt, Kk, n0 + in * 16 + r16, k0, lane);
#pragma unroll
        for (int im = 0; im < WMF; ++im)
#pragma unroll
            for (int in = 0; in < WNF; ++in)
                acc[im][in] = __builtin_amdgcn_wmma_f32_16x16x32_bf16(
                    false, af[im], false, bf[in], (short)0, acc[im][in],
                    false, false);
    }

    int mhalf = (lane >> 4) << 3;            // rows r..r+7 / +8 per C layout
#pragma unroll
    for (int im = 0; im < WMF; ++im) {
#pragma unroll
        for (int in = 0; in < WNF; ++in) {
            int col = n0 + in * 16 + r16;
#pragma unroll
            for (int r = 0; r < 8; ++r) {
                int m = m0 + im * 16 + r + mhalf;
                if (mode == 0) {
                    C[(size_t)m * Nn + col] = acc[im][in][r];
                } else {
                    int b = m >> 11, l = m & (LL - 1);
                    C[((size_t)b * J2 + col) * LL + l] = acc[im][in][r];
                }
            }
        }
    }
}

// ======= simple GEMM (one 16x16 tile / wave) for the small N=80 x_proj =====
__global__ void gemm_bf16_wmma_kernel(const __bf16* __restrict__ A,
                                      const __bf16* __restrict__ Bt,
                                      float* __restrict__ C,
                                      int M, int Nn, int Kk) {
    int lane = threadIdx.x & 31;
    int wave = threadIdx.x >> 5;
    int tile = blockIdx.x * 8 + wave;
    int ntn  = Nn >> 4;
    int tm   = tile / ntn;
    int tn   = tile - tm * ntn;
    if (tm >= (M >> 4)) return;              // whole-wave uniform exit

    int row = (tm << 4) + (lane & 15);
    int col = (tn << 4) + (lane & 15);

    v8f acc = {};
    for (int k0 = 0; k0 < Kk; k0 += 32) {
        v16bf a = load_frag_a(A, Kk, row, k0, lane);
        v16bf b = load_frag_b(Bt, Kk, col, k0, lane);
        acc = __builtin_amdgcn_wmma_f32_16x16x32_bf16(
            false, a, false, b, (short)0, acc, false, false);
    }

    int mhalf = (lane >> 4) << 3;
#pragma unroll
    for (int r = 0; r < 8; ++r) {
        int m = (tm << 4) + r + mhalf;
        C[(size_t)m * Nn + col] = acc[r];
    }
}

// ---------------- depthwise causal conv (K=4) + bias + SiLU ----------------
__global__ void conv_silu_kernel(const float* __restrict__ proj,
                                 const float* __restrict__ conv_w,
                                 const float* __restrict__ conv_b,
                                 float* __restrict__ hidden,
                                 __bf16* __restrict__ hbf) {
    int idx = blockIdx.x * 256 + threadIdx.x;      // over B*I*L
    int l  = idx & (LL - 1);
    int bi = idx >> 11;
    int i  = bi % II;
    int b  = bi / II;
    const float* row = proj + ((size_t)b * J2 + i) * LL;
    float s = conv_b[i];
#pragma unroll
    for (int k = 0; k < KK; ++k) {
        int ls = l - (KK - 1) + k;
        if (ls >= 0) s += row[ls] * conv_w[i * KK + k];
    }
    float h = s / (1.f + __expf(-s));
    hidden[((size_t)b * II + i) * LL + l] = h;
    hbf[((size_t)b * LL + l) * II + i]   = (__bf16)h;
}

// ---------------- dt = softplus(ts @ dt_w + dt_b) (K=48, VALU) -------------
__global__ void dt_softplus_kernel(const float* __restrict__ ssm_p,
                                   const float* __restrict__ dtw,
                                   const float* __restrict__ dtb,
                                   float* __restrict__ dtv) {
    int idx = blockIdx.x * 256 + threadIdx.x;      // over ML*I
    int i = idx % II;
    int m = idx / II;
    const float* ts = ssm_p + (size_t)m * 80;
    float s = dtb[i];
#pragma unroll 8
    for (int r = 0; r < DTR; ++r) s += ts[r] * dtw[r * II + i];
    dtv[idx] = (s > 20.f) ? s : log1pf(__expf(s));
}

// ---------------- selective scan, fused D-term + gate-SiLU + bf16 repack ---
// 128 threads = 8 channels x 16 states; state lives in a register across L.
__global__ void scan_kernel(const float* __restrict__ dtv,
                            const float* __restrict__ ssm_p,
                            const float* __restrict__ hidden,
                            const float* __restrict__ proj,
                            const float* __restrict__ A_log,
                            const float* __restrict__ Dp,
                            __bf16* __restrict__ ybf) {
    int n  = threadIdx.x & 15;
    int ii = threadIdx.x >> 4;               // 0..7
    int b  = blockIdx.x / (II / 8);
    int ig = blockIdx.x % (II / 8);
    int i  = ig * 8 + ii;

    float a = -__expf(A_log[i * NN + n]);
    float d = Dp[i];
    const float* hrow = hidden + ((size_t)b * II + i) * LL;
    const float* grow = proj + ((size_t)b * J2 + II + i) * LL;

    float st = 0.f;
    for (int l = 0; l < LL; ++l) {
        size_t m = (size_t)b * LL + l;
        float dt = dtv[m * II + i];
        float Bn = ssm_p[m * 80 + DTR + n];
        float Cn = ssm_p[m * 80 + DTR + NN + n];
        float u  = hrow[l];
        st = __expf(a * dt) * st + dt * Bn * u;
        float v = st * Cn;
        v += __shfl_xor(v, 8, 32);           // reduce over n (16-lane groups,
        v += __shfl_xor(v, 4, 32);           //  masks keep bit4 -> stay in group)
        v += __shfl_xor(v, 2, 32);
        v += __shfl_xor(v, 1, 32);
        if (n == 0) {
            float y = v + u * d;
            float g = grow[l];
            y *= g / (1.f + __expf(-g));     // * silu(gate)
            ybf[m * II + i] = (__bf16)y;
        }
    }
}

// ---------------- launch ----------------
extern "C" void kernel_launch(void* const* d_in, const int* in_sizes, int n_in,
                              void* d_out, int out_size, void* d_ws, size_t ws_size,
                              hipStream_t stream) {
    const float* input    = (const float*)d_in[0];
    const float* in_w     = (const float*)d_in[1];
    const float* conv_w   = (const float*)d_in[2];
    const float* conv_b   = (const float*)d_in[3];
    const float* xproj_w  = (const float*)d_in[4];
    const float* dt_w     = (const float*)d_in[5];
    const float* dt_b     = (const float*)d_in[6];
    const float* A_log    = (const float*)d_in[7];
    const float* Dp       = (const float*)d_in[8];
    const float* out_w    = (const float*)d_in[9];
    float* out = (float*)d_out;

    char* ws = (char*)d_ws;
    __bf16* Abf  = (__bf16*)(ws + OFF_ABF);
    __bf16* W1t  = (__bf16*)(ws + OFF_W1T);
    __bf16* Xt   = (__bf16*)(ws + OFF_XT);
    __bf16* Ot   = (__bf16*)(ws + OFF_OT);
    float*  projf = (float*)(ws + OFF_PROJ);
    float*  hid   = (float*)(ws + OFF_HID);
    __bf16* hbf  = (__bf16*)(ws + OFF_HBF);
    float*  ssm   = (float*)(ws + OFF_SSM);
    float*  dtv   = (float*)(ws + OFF_DTV);
    __bf16* ybf  = (__bf16*)(ws + OFF_YBF);

    // prep: bf16 convert + weight transposes
    cvt_bf16_kernel<<<(ML * HH) / 256, 256, 0, stream>>>(input, Abf, ML * HH);
    cvt_transpose_kernel<<<(HH * J2) / 256, 256, 0, stream>>>(in_w, W1t, HH, J2);
    cvt_transpose_kernel<<<(II * 80) / 256, 256, 0, stream>>>(xproj_w, Xt, II, 80);
    cvt_transpose_kernel<<<(II * HH) / 256, 256, 0, stream>>>(out_w, Ot, II, HH);

    // in_proj: [4096x768]x[768x3072] -> proj[b][j][l]   (blocked WMMA)
    gemm_bf16_wmma_blocked<<<(ML / 64) * (J2 / 256), 256, 0, stream>>>(
        Abf, W1t, projf, ML, J2, HH, 1);

    // depthwise conv + SiLU
    conv_silu_kernel<<<(BB * II * LL) / 256, 256, 0, stream>>>(
        projf, conv_w, conv_b, hid, hbf);

    // x_proj: [4096x1536]x[1536x80] -> ssm_p[m][80]   (simple WMMA, N=80)
    gemm_bf16_wmma_kernel<<<(ML / 16) * (80 / 16) / 8, 256, 0, stream>>>(
        hbf, Xt, ssm, ML, 80, II);

    // dt projection + softplus (K=48, VALU)
    dt_softplus_kernel<<<(ML * II) / 256, 256, 0, stream>>>(ssm, dt_w, dt_b, dtv);

    // selective scan + fused epilogue
    scan_kernel<<<BB * (II / 8), 128, 0, stream>>>(
        dtv, ssm, hid, projf, A_log, Dp, ybf);

    // out_proj: [4096x1536]x[1536x768] -> out[m][768]   (blocked WMMA)
    gemm_bf16_wmma_blocked<<<(ML / 64) * (HH / 256), 256, 0, stream>>>(
        ybf, Ot, out, ML, HH, II, 0);
}
